// GraphTransformerEncoder_46480136077661
// MI455X (gfx1250) — matrix operand order repaired and checked by reading, not compiled, not run on hardware
//
#include <hip/hip_runtime.h>
#include <hip/hip_bf16.h>
#include <cstdint>
#include <cstddef>

// ---------------------------------------------------------------------------
// GraphTransformerEncoder for MI455X (gfx1250, wave32, WMMA)
//   B=8 N=4096 D=128 H=4 C=32 L=3 DEG=16 F=512, nseg = 32768 nodes
// GEMMs: v_wmma_f32_16x16x32_bf16, 64x32 register-blocked per wave
// Attention fused per-node (fixed DEG=16 consecutive-edge grouping).
// ---------------------------------------------------------------------------

typedef __bf16 bf16;
typedef __attribute__((ext_vector_type(16))) __bf16 v16bf;
typedef __attribute__((ext_vector_type(4)))  __bf16 bf16x4;
typedef __attribute__((ext_vector_type(8)))  float  v8f;

#define NSEG   32768
#define DDIM   128
#define HEADS  4
#define CPH    32
#define FDIM   512
#define DEG    16
#define LAYERS 3

union FragBF16 { v16bf v; uint4 u[2]; };

// ---- weight transpose + f32->bf16: out[n*K + k] = bf16(in[k*N + n]) --------
__global__ void k_wt(const float* __restrict__ in, bf16* __restrict__ out,
                     int K, int N) {
  int id = blockIdx.x * blockDim.x + threadIdx.x;
  if (id >= K * N) return;
  int k = id % K;
  int n = id / K;
  out[id] = (bf16)in[(size_t)k * N + n];
}

// ---- elementwise f32 -> bf16 ----------------------------------------------
__global__ void k_f2b(const float* __restrict__ in, bf16* __restrict__ out, int n) {
  int id = blockIdx.x * blockDim.x + threadIdx.x;
  if (id < n) out[id] = (bf16)in[id];
}

// ---- WMMA GEMM: out[M,N] = A[M,K](bf16, row-major) x Bt[N,K](bf16)^T + bias
//   Each wave computes a 64x32 output tile: 4 M-subtiles x 2 N-subtiles,
//   8 independent accumulator chains, 8 WMMAs per 12 b128 loads per K-step.
//   act==1 -> SiLU.  outF (f32) and/or outB (bf16) epilogue targets.
//   4 waves (128 threads) per block. Requires M%256==0, N%32==0, K%32==0.
__global__ __launch_bounds__(128)
void k_gemm(const bf16* __restrict__ A, const bf16* __restrict__ Bt,
            const float* __restrict__ bias,
            float* __restrict__ outF, bf16* __restrict__ outB,
            int M, int N, int K, int act) {
  const int lane  = threadIdx.x & 31;
  const int wave  = threadIdx.x >> 5;
  const int mBase = (blockIdx.y * 4 + wave) * 64;
  const int nBase = blockIdx.x * 32;
  const int half  = lane >> 4;
  const int lm    = lane & 15;

  // A 16-bit frag layout: lanes 0-15 -> K {0..7,16..23}, lanes 16-31 -> K {8..15,24..31}
  const bf16* aRow0 = A + (size_t)(mBase + 0 * 16 + lm) * K + half * 8;
  const bf16* aRow1 = A + (size_t)(mBase + 1 * 16 + lm) * K + half * 8;
  const bf16* aRow2 = A + (size_t)(mBase + 2 * 16 + lm) * K + half * 8;
  const bf16* aRow3 = A + (size_t)(mBase + 3 * 16 + lm) * K + half * 8;
  // B 16-bit frag layout: lanes 0-15 -> K {0..7,16..23}, lanes 16-31 -> K {8..15,24..31} (col = lm)
  const bf16* bRow0 = Bt + (size_t)(nBase + 0 * 16 + lm) * K + half * 8;
  const bf16* bRow1 = Bt + (size_t)(nBase + 1 * 16 + lm) * K + half * 8;

  v8f c00 = {}, c01 = {}, c10 = {}, c11 = {};
  v8f c20 = {}, c21 = {}, c30 = {}, c31 = {};

  for (int k0 = 0; k0 < K; k0 += 32) {
    FragBF16 a0, a1, a2, a3, b0, b1;
    a0.u[0] = *(const uint4*)(aRow0 + k0);  a0.u[1] = *(const uint4*)(aRow0 + k0 + 16);
    a1.u[0] = *(const uint4*)(aRow1 + k0);  a1.u[1] = *(const uint4*)(aRow1 + k0 + 16);
    a2.u[0] = *(const uint4*)(aRow2 + k0);  a2.u[1] = *(const uint4*)(aRow2 + k0 + 16);
    a3.u[0] = *(const uint4*)(aRow3 + k0);  a3.u[1] = *(const uint4*)(aRow3 + k0 + 16);
    b0.u[0] = *(const uint4*)(bRow0 + k0);  b0.u[1] = *(const uint4*)(bRow0 + k0 + 16);
    b1.u[0] = *(const uint4*)(bRow1 + k0);  b1.u[1] = *(const uint4*)(bRow1 + k0 + 16);

    c00 = __builtin_amdgcn_wmma_f32_16x16x32_bf16(false, a0.v, false, b0.v, (short)0, c00, false, false);
    c01 = __builtin_amdgcn_wmma_f32_16x16x32_bf16(false, a0.v, false, b1.v, (short)0, c01, false, false);
    c10 = __builtin_amdgcn_wmma_f32_16x16x32_bf16(false, a1.v, false, b0.v, (short)0, c10, false, false);
    c11 = __builtin_amdgcn_wmma_f32_16x16x32_bf16(false, a1.v, false, b1.v, (short)0, c11, false, false);
    c20 = __builtin_amdgcn_wmma_f32_16x16x32_bf16(false, a2.v, false, b0.v, (short)0, c20, false, false);
    c21 = __builtin_amdgcn_wmma_f32_16x16x32_bf16(false, a2.v, false, b1.v, (short)0, c21, false, false);
    c30 = __builtin_amdgcn_wmma_f32_16x16x32_bf16(false, a3.v, false, b0.v, (short)0, c30, false, false);
    c31 = __builtin_amdgcn_wmma_f32_16x16x32_bf16(false, a3.v, false, b1.v, (short)0, c31, false, false);
  }

  // ---- epilogue: C/D layout: VGPR r -> M = r + half*8, N = lm --------------
  const int n0 = nBase + 0 * 16 + lm;
  const int n1 = nBase + 1 * 16 + lm;
  const float bn0 = bias ? bias[n0] : 0.0f;
  const float bn1 = bias ? bias[n1] : 0.0f;

  v8f* acc[4][2] = {{&c00, &c01}, {&c10, &c11}, {&c20, &c21}, {&c30, &c31}};
#pragma unroll
  for (int i = 0; i < 4; ++i) {
#pragma unroll
    for (int r = 0; r < 8; ++r) {
      const int m = mBase + i * 16 + r + half * 8;
      float v0 = (*acc[i][0])[r] + bn0;
      float v1 = (*acc[i][1])[r] + bn1;
      if (act == 1) {
        v0 = v0 / (1.0f + __expf(-v0));
        v1 = v1 / (1.0f + __expf(-v1));
      }
      const size_t i0 = (size_t)m * N + n0;
      const size_t i1 = (size_t)m * N + n1;
      if (outF) { outF[i0] = v0; outF[i1] = v1; }
      if (outB) { outB[i0] = (bf16)v0; outB[i1] = (bf16)v1; }
    }
  }
}

// ---- fused GATv2 attention: one wave per node, DEG=16 consecutive edges ----
//   lane -> 4 channels (float4); 8 lanes share a head -> shfl_xor reduction.
__global__ __launch_bounds__(256)
void k_gat(const float* __restrict__ gl, const float* __restrict__ gr,
           const float* __restrict__ att, const float* __restrict__ bias,
           const int* __restrict__ src, float* __restrict__ xg) {
  const int lane = threadIdx.x & 31;
  const int wave = threadIdx.x >> 5;
  const int node = blockIdx.x * 8 + wave;
  const int ch   = lane * 4;
  const int h    = ch >> 5;

  const float4 grv = *(const float4*)(gr   + (size_t)node * DDIM + ch);
  const float4 atv = *(const float4*)(att  + h * CPH + (ch & 31));
  const float4 bsv = *(const float4*)(bias + ch);

  const int* __restrict__ srcs = src + node * DEG;
  float lg[DEG];
#pragma unroll
  for (int d = 0; d < DEG; ++d) {
    const int s = srcs[d];
    if (d + 1 < DEG)
      __builtin_prefetch(gl + (size_t)srcs[d + 1] * DDIM + ch, 0, 1);
    float4 g = *(const float4*)(gl + (size_t)s * DDIM + ch);
    float4 e;
    e.x = g.x + grv.x; e.x = (e.x > 0.0f) ? e.x : 0.2f * e.x;
    e.y = g.y + grv.y; e.y = (e.y > 0.0f) ? e.y : 0.2f * e.y;
    e.z = g.z + grv.z; e.z = (e.z > 0.0f) ? e.z : 0.2f * e.z;
    e.w = g.w + grv.w; e.w = (e.w > 0.0f) ? e.w : 0.2f * e.w;
    float p = e.x * atv.x + e.y * atv.y + e.z * atv.z + e.w * atv.w;
    p += __shfl_xor(p, 1);
    p += __shfl_xor(p, 2);
    p += __shfl_xor(p, 4);   // all 8 lanes of this head now hold logit[d][h]
    lg[d] = p;
  }

  float m = lg[0];
#pragma unroll
  for (int d = 1; d < DEG; ++d) m = fmaxf(m, lg[d]);
  float den = 0.0f;
#pragma unroll
  for (int d = 0; d < DEG; ++d) { lg[d] = __expf(lg[d] - m); den += lg[d]; }
  const float inv = 1.0f / den;

  float4 acc = {0.0f, 0.0f, 0.0f, 0.0f};
#pragma unroll
  for (int d = 0; d < DEG; ++d) {
    const float a = lg[d] * inv;
    float4 g = *(const float4*)(gl + (size_t)srcs[d] * DDIM + ch);  // L2-resident
    acc.x += g.x * a; acc.y += g.y * a; acc.z += g.z * a; acc.w += g.w * a;
  }
  float4 o;
  o.x = acc.x + bsv.x; o.y = acc.y + bsv.y;
  o.z = acc.z + bsv.z; o.w = acc.w + bsv.w;
  *(float4*)(xg + (size_t)node * DDIM + ch) = o;
}

// ---- LayerNorm over D=128, one wave per node, bf16 output ------------------
__global__ __launch_bounds__(256)
void k_ln1(const float* __restrict__ x, const float* __restrict__ g,
           const float* __restrict__ b, bf16* __restrict__ out) {
  const int lane = threadIdx.x & 31;
  const int wave = threadIdx.x >> 5;
  const int node = blockIdx.x * 8 + wave;
  const int ch   = lane * 4;

  float4 v = *(const float4*)(x + (size_t)node * DDIM + ch);
  float  s = v.x + v.y + v.z + v.w;
  s += __shfl_xor(s, 1);  s += __shfl_xor(s, 2);  s += __shfl_xor(s, 4);
  s += __shfl_xor(s, 8);  s += __shfl_xor(s, 16);
  const float mean = s * (1.0f / 128.0f);
  float4 d;
  d.x = v.x - mean; d.y = v.y - mean; d.z = v.z - mean; d.w = v.w - mean;
  float q = d.x * d.x + d.y * d.y + d.z * d.z + d.w * d.w;
  q += __shfl_xor(q, 1);  q += __shfl_xor(q, 2);  q += __shfl_xor(q, 4);
  q += __shfl_xor(q, 8);  q += __shfl_xor(q, 16);
  const float rs = rsqrtf(q * (1.0f / 128.0f) + 1e-5f);
  float4 gv = *(const float4*)(g + ch);
  float4 bv = *(const float4*)(b + ch);
  bf16x4 o;
  o.x = (bf16)(d.x * rs * gv.x + bv.x);
  o.y = (bf16)(d.y * rs * gv.y + bv.y);
  o.z = (bf16)(d.z * rs * gv.z + bv.z);
  o.w = (bf16)(d.w * rs * gv.w + bv.w);
  *(bf16x4*)(out + (size_t)node * DDIM + ch) = o;
}

// ---- residual + LayerNorm: x = LN(xg + ho); writes bf16 (next layer) + opt f32
__global__ __launch_bounds__(256)
void k_ln2(const float* __restrict__ xg, const float* __restrict__ ho,
           const float* __restrict__ g, const float* __restrict__ b,
           bf16* __restrict__ outB, float* __restrict__ outF) {
  const int lane = threadIdx.x & 31;
  const int wave = threadIdx.x >> 5;
  const int node = blockIdx.x * 8 + wave;
  const int ch   = lane * 4;

  float4 a = *(const float4*)(xg + (size_t)node * DDIM + ch);
  float4 c = *(const float4*)(ho + (size_t)node * DDIM + ch);
  float4 v;
  v.x = a.x + c.x; v.y = a.y + c.y; v.z = a.z + c.z; v.w = a.w + c.w;
  float s = v.x + v.y + v.z + v.w;
  s += __shfl_xor(s, 1);  s += __shfl_xor(s, 2);  s += __shfl_xor(s, 4);
  s += __shfl_xor(s, 8);  s += __shfl_xor(s, 16);
  const float mean = s * (1.0f / 128.0f);
  float4 d;
  d.x = v.x - mean; d.y = v.y - mean; d.z = v.z - mean; d.w = v.w - mean;
  float q = d.x * d.x + d.y * d.y + d.z * d.z + d.w * d.w;
  q += __shfl_xor(q, 1);  q += __shfl_xor(q, 2);  q += __shfl_xor(q, 4);
  q += __shfl_xor(q, 8);  q += __shfl_xor(q, 16);
  const float rs = rsqrtf(q * (1.0f / 128.0f) + 1e-5f);
  float4 gv = *(const float4*)(g + ch);
  float4 bv = *(const float4*)(b + ch);
  float4 o;
  o.x = d.x * rs * gv.x + bv.x;
  o.y = d.y * rs * gv.y + bv.y;
  o.z = d.z * rs * gv.z + bv.z;
  o.w = d.w * rs * gv.w + bv.w;
  bf16x4 ob;
  ob.x = (bf16)o.x; ob.y = (bf16)o.y; ob.z = (bf16)o.z; ob.w = (bf16)o.w;
  *(bf16x4*)(outB + (size_t)node * DDIM + ch) = ob;
  if (outF) *(float4*)(outF + (size_t)node * DDIM + ch) = o;
}

// ---------------------------------------------------------------------------
extern "C" void kernel_launch(void* const* d_in, const int* in_sizes, int n_in,
                              void* d_out, int out_size, void* d_ws, size_t ws_size,
                              hipStream_t stream) {
  const float* x    = (const float*)d_in[0];
  const float* lw   = (const float*)d_in[1];
  const float* lb   = (const float*)d_in[2];
  const float* rw   = (const float*)d_in[3];
  const float* rb   = (const float*)d_in[4];
  const float* att  = (const float*)d_in[5];
  const float* gbia = (const float*)d_in[6];
  const float* l1g  = (const float*)d_in[7];
  const float* l1b  = (const float*)d_in[8];
  const float* f1w  = (const float*)d_in[9];
  const float* f1b  = (const float*)d_in[10];
  const float* f2w  = (const float*)d_in[11];
  const float* f2b  = (const float*)d_in[12];
  const float* l2g  = (const float*)d_in[13];
  const float* l2b  = (const float*)d_in[14];
  const int*   edge = (const int*)d_in[15];
  const int*   src  = edge;                  // row 0 of edge_index
  float*       out  = (float*)d_out;

  // --- carve workspace ------------------------------------------------------
  char*  ws  = (char*)d_ws;
  size_t off = 0;
  auto take = [&](size_t bytes) -> char* {
    char* p = ws + off;
    off += (bytes + 255) & ~(size_t)255;
    return p;
  };
  bf16*  xb   = (bf16*) take((size_t)NSEG * DDIM * sizeof(bf16));   //  8 MB
  float* gl   = (float*)take((size_t)NSEG * DDIM * sizeof(float));  // 16 MB
  float* gr   = (float*)take((size_t)NSEG * DDIM * sizeof(float));  // 16 MB
  float* xg   = (float*)take((size_t)NSEG * DDIM * sizeof(float));  // 16 MB
  bf16*  hb   = (bf16*) take((size_t)NSEG * DDIM * sizeof(bf16));   //  8 MB
  bf16*  hidb = (bf16*) take((size_t)NSEG * FDIM * sizeof(bf16));   // 32 MB
  float* ho   = (float*)take((size_t)NSEG * DDIM * sizeof(float));  // 16 MB
  bf16*  wlT  = (bf16*) take((size_t)LAYERS * DDIM * DDIM * sizeof(bf16));
  bf16*  wrT  = (bf16*) take((size_t)LAYERS * DDIM * DDIM * sizeof(bf16));
  bf16*  f1T  = (bf16*) take((size_t)LAYERS * DDIM * FDIM * sizeof(bf16));
  bf16*  f2T  = (bf16*) take((size_t)LAYERS * DDIM * FDIM * sizeof(bf16));

  // --- weight prep: transpose to [N][K] bf16 --------------------------------
  for (int l = 0; l < LAYERS; ++l) {
    k_wt<<<(DDIM * DDIM + 255) / 256, 256, 0, stream>>>(
        lw + (size_t)l * DDIM * DDIM, wlT + (size_t)l * DDIM * DDIM, DDIM, DDIM);
    k_wt<<<(DDIM * DDIM + 255) / 256, 256, 0, stream>>>(
        rw + (size_t)l * DDIM * DDIM, wrT + (size_t)l * DDIM * DDIM, DDIM, DDIM);
    k_wt<<<(DDIM * FDIM + 255) / 256, 256, 0, stream>>>(
        f1w + (size_t)l * DDIM * FDIM, f1T + (size_t)l * DDIM * FDIM, DDIM, FDIM);
    k_wt<<<(DDIM * FDIM + 255) / 256, 256, 0, stream>>>(
        f2w + (size_t)l * DDIM * FDIM, f2T + (size_t)l * DDIM * FDIM, FDIM, DDIM);
  }
  k_f2b<<<(NSEG * DDIM + 255) / 256, 256, 0, stream>>>(x, xb, NSEG * DDIM);

  // --- layers ---------------------------------------------------------------
  const dim3 blk(128);
  const dim3 gD (DDIM / 32, NSEG / 256);   // 64x32 tile per wave, 4 waves/block
  const dim3 gF (FDIM / 32, NSEG / 256);
  for (int l = 0; l < LAYERS; ++l) {
    k_gemm<<<gD, blk, 0, stream>>>(xb, wlT + (size_t)l * DDIM * DDIM,
                                   lb + l * DDIM, gl, nullptr,
                                   NSEG, DDIM, DDIM, 0);
    k_gemm<<<gD, blk, 0, stream>>>(xb, wrT + (size_t)l * DDIM * DDIM,
                                   rb + l * DDIM, gr, nullptr,
                                   NSEG, DDIM, DDIM, 0);
    k_gat<<<NSEG / 8, 256, 0, stream>>>(gl, gr, att + l * HEADS * CPH,
                                        gbia + l * DDIM, src, xg);
    k_ln1<<<NSEG / 8, 256, 0, stream>>>(xg, l1g + l * DDIM, l1b + l * DDIM, hb);
    k_gemm<<<gF, blk, 0, stream>>>(hb, f1T + (size_t)l * DDIM * FDIM,
                                   f1b + l * FDIM, nullptr, hidb,
                                   NSEG, FDIM, DDIM, 1);          // SiLU -> bf16
    k_gemm<<<gD, blk, 0, stream>>>(hidb, f2T + (size_t)l * DDIM * FDIM,
                                   f2b + l * DDIM, ho, nullptr,
                                   NSEG, DDIM, FDIM, 0);
    float* fOut = (l == LAYERS - 1) ? out : nullptr;
    k_ln2<<<NSEG / 8, 256, 0, stream>>>(xg, ho, l2g + l * DDIM, l2b + l * DDIM,
                                        xb, fOut);
  }
}
